// AttentionEncoder_63599875719263
// MI455X (gfx1250) — matrix operand order repaired and checked by reading, not compiled, not run on hardware
//
#include <hip/hip_runtime.h>

// ---------------------------------------------------------------------------
// Problem constants
// ---------------------------------------------------------------------------
constexpr int BB    = 64;    // batch
constexpr int LSEQ  = 512;   // input tokens
constexpr int EMB   = 128;   // embedding dim
constexpr int CCH   = 128;   // conv out channels (per kernel size)
constexpr int DM    = 384;   // d_model
constexpr int NH    = 4;     // heads
constexpr int DK    = 96;    // head dim
constexpr int LP    = 513;   // conv / attention sequence length
constexpr int LPAD  = 544;   // padded seq (34*16 = 17*32)
constexpr int EPL   = 520;   // padded embedding rows per batch (conv halo of 4)
constexpr int PT    = 33;    // p-tiles covering 0..527 (>=513)
constexpr int KT    = 34;    // k-tiles covering 0..543
constexpr float QSCALE = 0.10206207261596575f;   // 1/sqrt(96)

// conv packed-weight offsets (in halves)
constexpr int WOFS4 = 0;
constexpr int WOFS6 = 128 * 512;             // 65536
constexpr int WOFS8 = WOFS6 + 128 * 768;     // 163840
constexpr int WCONV_TOTAL = WOFS8 + 128 * 1024;  // 294912 halves

// ---------------------------------------------------------------------------
// WMMA helpers (CDNA5 gfx1250, wave32)
// ---------------------------------------------------------------------------
typedef __attribute__((ext_vector_type(16))) _Float16 v16h;
typedef __attribute__((ext_vector_type(8)))  float    v8f;
typedef __attribute__((ext_vector_type(4)))  float    vf4;

union VH16 { v16h v; _Float16 h[16]; vf4 q[2]; };

static __device__ inline v8f wmma32(v16h a, v16h b, v8f c) {
  return __builtin_amdgcn_wmma_f32_16x16x32_f16(
      /*neg_a=*/false, a, /*neg_b=*/false, b,
      /*c_mod=*/(short)0, c, /*reuse_a=*/false, /*reuse_b=*/false);
}

static __device__ inline v8f acc_zero() {
  v8f z;
#pragma unroll
  for (int i = 0; i < 8; ++i) z[i] = 0.0f;
  return z;
}

// A fragment, K contiguous in memory at rowbase[k0 + ...]
// lane<16 : K = k0 + {0..7, 16..23};  lane>=16 : K = k0 + {8..15, 24..31}
static __device__ inline v16h load_a_contig(const _Float16* rowbase, int k0, int khalf) {
  const _Float16* p = rowbase + k0 + khalf * 8;
  VH16 u;
  u.q[0] = *(const vf4*)(p);
  u.q[1] = *(const vf4*)(p + 16);
  return u.v;
}

// B fragment: lane holds column N = lane%16, K = k0 + 8*khalf + {0..7, 16..23}.
// colbase points at column's K-major row.
static __device__ inline v16h load_b_frag(const _Float16* colbase, int k0, int khalf) {
  const _Float16* p = colbase + k0 + khalf * 8;
  VH16 u;
  u.q[0] = *(const vf4*)(p);
  u.q[1] = *(const vf4*)(p + 16);
  return u.v;
}

// ---------------------------------------------------------------------------
// Kernel 0: zero fill
// ---------------------------------------------------------------------------
__global__ void zero_fill(unsigned int* p, size_t nwords) {
  size_t i = (size_t)blockIdx.x * blockDim.x + threadIdx.x;
  size_t stride = (size_t)gridDim.x * blockDim.x;
  for (; i < nwords; i += stride) p[i] = 0u;
}

// ---------------------------------------------------------------------------
// Kernel 1: pack conv weights (C,E,k) f32 -> wB[c][t*128+e] f16
// ---------------------------------------------------------------------------
__global__ void pack_conv(const float* w4, const float* w6, const float* w8,
                          _Float16* wB) {
  int g = blockIdx.x * blockDim.x + threadIdx.x;
  if (g >= WCONV_TOTAL) return;
  const float* src; int ktot, ksz, rel;
  if (g < WOFS6)      { src = w4; ktot = 512;  ksz = 4; rel = g; }
  else if (g < WOFS8) { src = w6; ktot = 768;  ksz = 6; rel = g - WOFS6; }
  else                { src = w8; ktot = 1024; ksz = 8; rel = g - WOFS8; }
  int c = rel / ktot, kk = rel % ktot;
  int t = kk >> 7, e = kk & 127;
  wB[g] = (_Float16)src[(c * 128 + e) * ksz + t];
}

// ---------------------------------------------------------------------------
// Kernel 2: pack Wq/Wk/Wv/Wo (D,D) f32 -> Wt[which][n][k] f16 (transposed)
// ---------------------------------------------------------------------------
__global__ void pack_wt(const float* Wq, const float* Wk, const float* Wv,
                        const float* Wo, _Float16* Wt) {
  int g = blockIdx.x * blockDim.x + threadIdx.x;
  if (g >= 4 * DM * DM) return;
  int which = g / (DM * DM), rel = g % (DM * DM);
  int n = rel / DM, kk = rel % DM;
  const float* src = (which == 0) ? Wq : (which == 1) ? Wk : (which == 2) ? Wv : Wo;
  Wt[g] = (_Float16)src[kk * DM + n];
}

// ---------------------------------------------------------------------------
// Kernel 3: embedding gather into zero-padded ep (B, EPL, EMB) f16
// ---------------------------------------------------------------------------
__global__ void embed_gather(const int* x, const float* emb, _Float16* ep) {
  int blk = blockIdx.x;                 // b*LSEQ + l
  int b = blk / LSEQ, l = blk % LSEQ;
  int tok = x[b * LSEQ + l];
  int e = threadIdx.x;
  ep[((size_t)(b * EPL + l + 4)) * EMB + e] = (_Float16)emb[tok * EMB + e];
}

// ---------------------------------------------------------------------------
// Kernel 4: conv1d as implicit GEMM via WMMA; bias + ReLU -> xs f16 (B,LPAD,DM)
// ---------------------------------------------------------------------------
__global__ __launch_bounds__(32) void conv_wmma(
    const _Float16* ep, const _Float16* wB,
    const float* b4, const float* b6, const float* b8, _Float16* xs) {
  int bp = blockIdx.x; int b = bp / PT; int pt = bp % PT; int p0 = pt * 16;
  int y = blockIdx.y;  int conv = y >> 3; int c0 = (y & 7) * 16;
  int ksz  = (conv == 0) ? 4 : (conv == 1) ? 6 : 8;
  int ktot = ksz * 128;
  int wofs = (conv == 0) ? WOFS4 : (conv == 1) ? WOFS6 : WOFS8;
  const float* bias = (conv == 0) ? b4 : (conv == 1) ? b6 : b8;
  int pad = ksz >> 1;

  int lane  = threadIdx.x & 31;
  int l15   = lane & 15;
  int khalf = lane >> 4;

  const _Float16* wcol = wB + wofs + (size_t)(c0 + l15) * ktot;
  v8f acc = acc_zero();

  for (int k0 = 0; k0 < ktot; k0 += 32) {
    int kbase = k0 + khalf * 8;
    int t = kbase >> 7, e = kbase & 127;
    int row = p0 + l15 + t - pad + 4;
    row = row < 0 ? 0 : (row > EPL - 1 ? EPL - 1 : row);
    const _Float16* pa = ep + ((size_t)(b * EPL + row)) * EMB + e;
    VH16 ua;
    ua.q[0] = *(const vf4*)(pa);
    ua.q[1] = *(const vf4*)(pa + 16);
    v16h wf = load_b_frag(wcol, k0, khalf);
    acc = wmma32(ua.v, wf, acc);
  }

  int cabs = conv * CCH + c0 + l15;
  float bv = bias[c0 + l15];
#pragma unroll
  for (int i = 0; i < 8; ++i) {
    int p = p0 + khalf * 8 + i;
    if (p < LP) {
      float v = acc[i] + bv;
      v = v > 0.0f ? v : 0.0f;
      xs[((size_t)(b * LPAD + p)) * DM + cabs] = (_Float16)v;
    }
  }
}

// ---------------------------------------------------------------------------
// Kernel 5: QKV projections via WMMA.
//   Q,K: (B,H,LPAD,DK) f16 ; V stored transposed VT: (B,H,DK,LPAD) f16
// ---------------------------------------------------------------------------
__global__ __launch_bounds__(32) void qkv_wmma(
    const _Float16* xs, const _Float16* Wt,
    const float* bq, const float* bk, const float* bv,
    _Float16* Q, _Float16* Kb, _Float16* VT) {
  int bp = blockIdx.x; int b = bp / PT; int pt = bp % PT; int p0 = pt * 16;
  int n0 = blockIdx.y * 16;
  int proj = blockIdx.z;

  const _Float16* W = Wt + (size_t)proj * DM * DM;
  const float* bias = (proj == 0) ? bq : (proj == 1) ? bk : bv;

  int lane = threadIdx.x & 31, l15 = lane & 15, khalf = lane >> 4;
  const _Float16* arow = xs + ((size_t)(b * LPAD + p0 + l15)) * DM;
  const _Float16* bcol = W + (size_t)(n0 + l15) * DM;

  v8f acc = acc_zero();
  for (int k0 = 0; k0 < DM; k0 += 32)
    acc = wmma32(load_a_contig(arow, k0, khalf), load_b_frag(bcol, k0, khalf), acc);

  int n = n0 + l15, hd = n / DK, dk = n % DK;
  float bb = bias[n];
  float sc = (proj == 0) ? QSCALE : 1.0f;
#pragma unroll
  for (int i = 0; i < 8; ++i) {
    int p = p0 + khalf * 8 + i;
    float v = (acc[i] + bb) * sc;
    if (proj == 2)
      VT[(((size_t)(b * NH + hd)) * DK + dk) * LPAD + p] = (_Float16)v;
    else {
      _Float16* dst = (proj == 0) ? Q : Kb;
      dst[(((size_t)(b * NH + hd)) * LPAD + p) * DK + dk] = (_Float16)v;
    }
  }
}

// ---------------------------------------------------------------------------
// Kernel 6: attention per (b, h, q-tile): WMMA scores -> LDS softmax ->
//           column sums (pool weights) -> WMMA P@V -> ctx f16 (B,LPAD,DM)
// ---------------------------------------------------------------------------
__global__ __launch_bounds__(128) void attn_kernel(
    const _Float16* Q, const _Float16* Kb, const _Float16* VT,
    _Float16* ctx, float* colsum) {
  __shared__ float S[16][LPAD];     // 34 KB

  int bid = blockIdx.x;
  int qt = bid % PT; int bh = bid / PT;
  int hd = bh % NH;  int b = bh / NH;
  int q0 = qt * 16;

  int tid = threadIdx.x, wave = tid >> 5, lane = tid & 31;
  int l15 = lane & 15, khalf = lane >> 4;

  const _Float16* Qbase = Q  + ((size_t)(b * NH + hd)) * LPAD * DK;
  const _Float16* Kbase = Kb + ((size_t)(b * NH + hd)) * LPAD * DK;

  // ---- scores S = Q K^T (Q pre-scaled) ----
  for (int kt = wave; kt < KT; kt += 4) {
    v8f acc = acc_zero();
    const _Float16* arow = Qbase + (size_t)(q0 + l15) * DK;
    const _Float16* bcol = Kbase + (size_t)(kt * 16 + l15) * DK;
    for (int k0 = 0; k0 < DK; k0 += 32)
      acc = wmma32(load_a_contig(arow, k0, khalf), load_b_frag(bcol, k0, khalf), acc);
    int col = kt * 16 + l15;
#pragma unroll
    for (int i = 0; i < 8; ++i) {
      int m = khalf * 8 + i;
      S[m][col] = (col < LP) ? acc[i] : -1e30f;
    }
  }
  __syncthreads();

  // ---- per-row softmax ----
  if (tid < 16) {
    float mx = -1e30f;
    for (int c = 0; c < LPAD; ++c) mx = fmaxf(mx, S[tid][c]);
    float s = 0.0f;
    for (int c = 0; c < LPAD; ++c) { float e = __expf(S[tid][c] - mx); S[tid][c] = e; s += e; }
    float inv = 1.0f / s;
    for (int c = 0; c < LPAD; ++c) S[tid][c] *= inv;
  }
  __syncthreads();

  // ---- pooling weights: column sums over valid q rows (summed over h via atomics)
  int qv = LP - q0; if (qv > 16) qv = 16;
  for (int c = tid; c < LP; c += 128) {
    float s = 0.0f;
    for (int r = 0; r < qv; ++r) s += S[r][c];
    atomicAdd(&colsum[b * LPAD + c], s);
  }

  // ---- context = P @ V ----
  const _Float16* Vbase = VT + ((size_t)(b * NH + hd)) * DK * LPAD;
  for (int nt = wave; nt < 6; nt += 4) {
    int n0 = nt * 16;
    v8f acc = acc_zero();
    for (int k0 = 0; k0 < LPAD; k0 += 32) {
      VH16 ua;
      int kb = k0 + khalf * 8;
#pragma unroll
      for (int j = 0; j < 8; ++j) {
        ua.h[j]     = (_Float16)S[l15][kb + j];
        ua.h[8 + j] = (_Float16)S[l15][kb + 16 + j];
      }
      v16h vf = load_b_frag(Vbase + (size_t)(n0 + l15) * LPAD, k0, khalf);
      acc = wmma32(ua.v, vf, acc);
    }
#pragma unroll
    for (int i = 0; i < 8; ++i) {
      int p = q0 + khalf * 8 + i;
      if (p < LP)
        ctx[((size_t)(b * LPAD + p)) * DM + hd * DK + n0 + l15] = (_Float16)acc[i];
    }
  }
}

// ---------------------------------------------------------------------------
// Kernel 7: per-batch sum of colsum (pool-weight denominator)
// ---------------------------------------------------------------------------
__global__ void wsum_kernel(const float* colsum, float* wsum) {
  __shared__ float red[256];
  int b = blockIdx.x, tid = threadIdx.x;
  float s = 0.0f;
  for (int c = tid; c < LP; c += 256) s += colsum[b * LPAD + c];
  red[tid] = s; __syncthreads();
  for (int st = 128; st > 0; st >>= 1) {
    if (tid < st) red[tid] += red[tid + st];
    __syncthreads();
  }
  if (tid == 0) wsum[b] = red[0];
}

// ---------------------------------------------------------------------------
// Kernel 8: output proj (WMMA) + bias + residual + LayerNorm + weighted pool
// ---------------------------------------------------------------------------
__global__ __launch_bounds__(256) void oproj_ln_pool(
    const _Float16* ctx, const _Float16* Wt, const float* bo,
    const _Float16* xs, const float* gamma, const float* beta,
    const float* colsum, const float* wsum, float* out) {
  __shared__ float Hs[16][DM];      // 24 KB

  int bp = blockIdx.x; int b = bp / PT; int pt = bp % PT; int p0 = pt * 16;
  int tid = threadIdx.x, wave = tid >> 5, lane = tid & 31;
  int l15 = lane & 15, khalf = lane >> 4;

  const _Float16* Wo = Wt + (size_t)3 * DM * DM;

  for (int nt = wave * 3; nt < wave * 3 + 3; ++nt) {
    int n0 = nt * 16;
    v8f acc = acc_zero();
    const _Float16* arow = ctx + ((size_t)(b * LPAD + p0 + l15)) * DM;
    const _Float16* bcol = Wo + (size_t)(n0 + l15) * DM;
    for (int k0 = 0; k0 < DM; k0 += 32)
      acc = wmma32(load_a_contig(arow, k0, khalf), load_b_frag(bcol, k0, khalf), acc);
    int n = n0 + l15;
    float bb = bo[n];
#pragma unroll
    for (int i = 0; i < 8; ++i) {
      int m = khalf * 8 + i;
      float res = (float)xs[((size_t)(b * LPAD + p0 + m)) * DM + n];
      Hs[m][n] = acc[i] + bb + res;
    }
  }
  __syncthreads();

  if (tid < 16) {
    int p = p0 + tid;
    if (p < LP) {
      float mu = 0.0f;
      for (int d = 0; d < DM; ++d) mu += Hs[tid][d];
      mu *= (1.0f / DM);
      float var = 0.0f;
      for (int d = 0; d < DM; ++d) { float dd = Hs[tid][d] - mu; var += dd * dd; }
      var *= (1.0f / DM);
      float rs = rsqrtf(var + 1e-5f);
      const float invHL = 1.0f / (NH * (float)LP);
      float denom = wsum[b] * invHL + 1e-8f;
      float w = (colsum[b * LPAD + p] * invHL) / denom;
      for (int d = 0; d < DM; ++d) {
        float nv = (Hs[tid][d] - mu) * rs * gamma[d] + beta[d];
        atomicAdd(&out[b * DM + d], w * nv);
      }
    }
  }
}

// ---------------------------------------------------------------------------
// Host launcher
// ---------------------------------------------------------------------------
extern "C" void kernel_launch(void* const* d_in, const int* in_sizes, int n_in,
                              void* d_out, int out_size, void* d_ws, size_t ws_size,
                              hipStream_t stream) {
  const int*   x    = (const int*)  d_in[0];
  const float* emb  = (const float*)d_in[1];
  const float* w4   = (const float*)d_in[2];
  const float* b4   = (const float*)d_in[3];
  const float* w6   = (const float*)d_in[4];
  const float* b6   = (const float*)d_in[5];
  const float* w8   = (const float*)d_in[6];
  const float* b8   = (const float*)d_in[7];
  const float* Wq   = (const float*)d_in[8];
  const float* bq   = (const float*)d_in[9];
  const float* Wk   = (const float*)d_in[10];
  const float* bk   = (const float*)d_in[11];
  const float* Wv   = (const float*)d_in[12];
  const float* bv   = (const float*)d_in[13];
  const float* Wo   = (const float*)d_in[14];
  const float* bo   = (const float*)d_in[15];
  const float* gam  = (const float*)d_in[16];
  const float* bet  = (const float*)d_in[17];

  char* ws = (char*)d_ws;
  size_t off = 0;
  _Float16* ep   = (_Float16*)(ws + off); off += (size_t)BB * EPL * EMB * 2;          // 8.5 MB
  _Float16* wBc  = (_Float16*)(ws + off); off += (size_t)WCONV_TOTAL * 2;             // 0.6 MB
  _Float16* Wt   = (_Float16*)(ws + off); off += (size_t)4 * DM * DM * 2;             // 1.2 MB
  _Float16* xs   = (_Float16*)(ws + off); off += (size_t)BB * LPAD * DM * 2;          // 26.7 MB
  _Float16* Qb   = (_Float16*)(ws + off); off += (size_t)BB * NH * LPAD * DK * 2;
  _Float16* Kbf  = (_Float16*)(ws + off); off += (size_t)BB * NH * LPAD * DK * 2;
  _Float16* VT   = (_Float16*)(ws + off); off += (size_t)BB * NH * LPAD * DK * 2;
  _Float16* ctx  = (_Float16*)(ws + off); off += (size_t)BB * LPAD * DM * 2;
  float*    csum = (float*)   (ws + off); off += (size_t)BB * LPAD * 4;
  float*    wsum = (float*)   (ws + off); off += 256;
  size_t used_words = off / 4;

  // zero workspace (padding must be 0; colsum/pool are accumulated atomically)
  zero_fill<<<4096, 256, 0, stream>>>((unsigned int*)ws, used_words);
  zero_fill<<<64, 256, 0, stream>>>((unsigned int*)d_out, (size_t)out_size);

  pack_conv<<<(WCONV_TOTAL + 255) / 256, 256, 0, stream>>>(w4, w6, w8, wBc);
  pack_wt<<<(4 * DM * DM + 255) / 256, 256, 0, stream>>>(Wq, Wk, Wv, Wo, Wt);
  embed_gather<<<BB * LSEQ, EMB, 0, stream>>>(x, emb, ep);

  conv_wmma<<<dim3(BB * PT, 24), 32, 0, stream>>>(ep, wBc, b4, b6, b8, xs);
  qkv_wmma<<<dim3(BB * PT, 24, 3), 32, 0, stream>>>(xs, Wt, bq, bk, bv, Qb, Kbf, VT);
  attn_kernel<<<BB * NH * PT, 128, 0, stream>>>(Qb, Kbf, VT, ctx, csum);
  wsum_kernel<<<BB, 256, 0, stream>>>(csum, wsum);
  oproj_ln_pool<<<BB * PT, 256, 0, stream>>>(ctx, Wt, bo, xs, gam, bet, csum, wsum,
                                             (float*)d_out);
}
